// PerformerAttention_36687610643079
// MI455X (gfx1250) — matrix operand order repaired and checked by reading, not compiled, not run on hardware
//
#include <hip/hip_runtime.h>
#include <hip/hip_bf16.h>

// ---------------------------------------------------------------------------
// Performer attention for MI455X (gfx1250), wave32, WMMA f32_16x16x32_f16,
// async global->LDS staging (ASYNCcnt) with double buffering when available.
// B=4, S=4096, DIM=1024, H=16, HD=64, M=64.
// ---------------------------------------------------------------------------

typedef __attribute__((ext_vector_type(16))) _Float16 v16h;
typedef __attribute__((ext_vector_type(8)))  _Float16 v8h;
typedef __attribute__((ext_vector_type(8)))  float    v8f;
typedef __attribute__((__vector_size__(16))) int v4i;

#define S_LEN 4096
#define DIM   1024
#define NH    16
#define HD    64
#define NF    64   // num random features M

#if __has_builtin(__builtin_amdgcn_global_load_async_to_lds_b128) && \
    __has_builtin(__builtin_amdgcn_s_wait_asynccnt)
#define HAVE_ASYNC_LDS 1
#else
#define HAVE_ASYNC_LDS 0
#endif

#if HAVE_ASYNC_LDS
// One 16-byte async global->LDS copy per lane (GLOBAL_LOAD_ASYNC_TO_LDS_B128,
// tracked by ASYNCcnt). The builtin takes generic int4* pointers.
__device__ __forceinline__ void async_copy16(const _Float16* g, _Float16* l) {
  __builtin_amdgcn_global_load_async_to_lds_b128((v4i*)g, (v4i*)l, 0, 0);
}
#endif

__device__ __forceinline__ v16h combine16(v8h lo, v8h hi) {
  v16h a;
#pragma unroll
  for (int i = 0; i < 8; ++i) { a[i] = lo[i]; a[i + 8] = hi[i]; }
  return a;
}

// Output index mapping.
// LAYOUT 0: plain row-major [rows, ldc]
// LAYOUT 1: rows over (b,s) [b*4096+s], cols over (h,hd) -> store [b,h,s,hd]
// LAYOUT 2: rows over (b,h,s), cols = d -> store [b,s, h*64+d] (merged heads)
template <int LAYOUT>
__device__ __forceinline__ long out_index(long row, int col, int ldc) {
  if constexpr (LAYOUT == 0) {
    return row * (long)ldc + col;
  } else if constexpr (LAYOUT == 1) {
    long b = row >> 12, s = row & 4095;
    int h = col >> 6, hd = col & 63;
    return (((b * NH + h) << 12) + s) * HD + hd;
  } else {
    long bh = row >> 12, s = row & 4095;
    long b = bh >> 4, h = bh & 15;
    return ((b << 12) + s) * DIM + h * HD + col;
  }
}

// Generic tiled WMMA GEMM: C[rows, cols] = A[rows,K] * B[cols,K]^T (+bias)
// Block tile 128x64, BK=64, 256 threads = 8 waves (4x2), wave tile 32x32.
// mat_mod != 0 selects a per-head B matrix: B += ((row0/4096) % mat_mod)*64*K.
template <int LAYOUT, bool RELU, bool HAS_BIAS, bool OUT_HALF>
__global__ __launch_bounds__(256) void gemm_wmma(
    const _Float16* __restrict__ A, const _Float16* __restrict__ B,
    const float* __restrict__ bias, void* __restrict__ out,
    int K, int mat_mod) {
  constexpr int BM = 128, BN = 64, BK = 64;
  __shared__ _Float16 sA[2][BM * BK];
  __shared__ _Float16 sB[2][BN * BK];

  const int t = threadIdx.x;
  const int wave = t >> 5, lane = t & 31;
  const int r = lane & 15, hi = lane >> 4;
  const int wm = wave & 3, wn = wave >> 2;  // 4x2 wave grid
  const long row0 = (long)blockIdx.x * BM;
  const int col0 = blockIdx.y * BN;
  const int ldc = gridDim.y * BN;

  const _Float16* Bp = B;
  if (mat_mod) Bp += (size_t)((row0 >> 12) % mat_mod) * (size_t)BN * K;

  v8f acc[2][2] = {};

  // One BK-tile of WMMA work on the given LDS buffers.
  auto compute = [&](const _Float16* la, const _Float16* lb) {
#pragma unroll
    for (int ks = 0; ks < 2; ++ks) {  // two K=32 WMMA steps per BK
      v16h af[2], bf[2];
#pragma unroll
      for (int tm = 0; tm < 2; ++tm) {
        const _Float16* pa = &la[(wm * 32 + tm * 16 + r) * BK + ks * 32 + hi * 16];
        af[tm] = combine16(*(const v8h*)pa, *(const v8h*)(pa + 8));
      }
#pragma unroll
      for (int tn = 0; tn < 2; ++tn) {
        const _Float16* pb = &lb[(wn * 32 + tn * 16 + r) * BK + ks * 32 + hi * 16];
        bf[tn] = *(const v16h*)pb;
      }
#pragma unroll
      for (int tm = 0; tm < 2; ++tm)
#pragma unroll
        for (int tn = 0; tn < 2; ++tn)
          acc[tm][tn] = __builtin_amdgcn_wmma_f32_16x16x32_f16(
              false, af[tm], false, bf[tn], (short)0, acc[tm][tn], false, false);
    }
  };

#if HAVE_ASYNC_LDS
  // Async double-buffered pipeline: prefetch tile i+1 while computing tile i.
  auto stage = [&](int buf, int k0) {
#pragma unroll
    for (int p = 0; p < 4; ++p) {  // A tile: 128x64 halves, 16B per thread/pass
      int i = (p * 256 + t) * 8;
      int rr = i >> 6, cc = i & 63;
      async_copy16(&A[(row0 + rr) * (long)K + k0 + cc], &sA[buf][i]);
    }
#pragma unroll
    for (int p = 0; p < 2; ++p) {  // B tile: 64x64 halves
      int i = (p * 256 + t) * 8;
      int rr = i >> 6, cc = i & 63;
      async_copy16(&Bp[(long)(col0 + rr) * K + k0 + cc], &sB[buf][i]);
    }
  };
  const int nIter = K / BK;
  stage(0, 0);
  for (int it = 0; it < nIter; ++it) {
    __builtin_amdgcn_s_wait_asynccnt(0);  // this wave's copies have landed
    __syncthreads();                      // all waves landed + prev compute done
    if (it + 1 < nIter) stage((it + 1) & 1, (it + 1) * BK);
    compute(sA[it & 1], sB[it & 1]);
  }
#else
  // Fallback: synchronous single-buffer staging through VGPRs.
  for (int k0 = 0; k0 < K; k0 += BK) {
#pragma unroll
    for (int p = 0; p < 4; ++p) {
      int i = (p * 256 + t) * 8;
      int rr = i >> 6, cc = i & 63;
      *(v8h*)&sA[0][i] = *(const v8h*)&A[(row0 + rr) * (long)K + k0 + cc];
    }
#pragma unroll
    for (int p = 0; p < 2; ++p) {
      int i = (p * 256 + t) * 8;
      int rr = i >> 6, cc = i & 63;
      *(v8h*)&sB[0][i] = *(const v8h*)&Bp[(long)(col0 + rr) * K + k0 + cc];
    }
    __syncthreads();
    compute(sA[0], sB[0]);
    __syncthreads();
  }
#endif

  // Store: C layout -> VGPR v holds row (v + 8*hi), lane r holds column.
#pragma unroll
  for (int tm = 0; tm < 2; ++tm)
#pragma unroll
    for (int tn = 0; tn < 2; ++tn) {
      const int ccol = col0 + wn * 32 + tn * 16 + r;
      float bv = 0.0f;
      if (HAS_BIAS) bv = bias[ccol];
#pragma unroll
      for (int v = 0; v < 8; ++v) {
        long grow = row0 + wm * 32 + tm * 16 + hi * 8 + v;
        float val = acc[tm][tn][v] + bv;
        if (RELU) val = fmaxf(val, 0.0f) * 0.125f;  // relu * 1/sqrt(M=64)
        long idx = out_index<LAYOUT>(grow, ccol, ldc);
        if constexpr (OUT_HALF)
          ((_Float16*)out)[idx] = (_Float16)val;
        else
          ((float*)out)[idx] = val;
      }
    }
}

// kv kernel: per (b,h) block computes kvT[d][m] = sum_s v[s,d] * k'[s,m].
// A = v^T (16x32 over d,s), B = k' (32x16 over s,m, staged transposed so the
// per-lane fragment reads are contiguous). 8 waves: wave (wd, wm2) owns a
// 16(d) x 32(m) patch of the 64x64 result. Staging transposes in registers,
// so it keeps the synchronous path.
__global__ __launch_bounds__(256) void kv_wmma(
    const _Float16* __restrict__ kprime,  // [B*H, S, M]
    const _Float16* __restrict__ vmat,    // [B*H, S, HD]
    _Float16* __restrict__ kvT) {         // [B*H, HD, M]
  __shared__ _Float16 sV[64 * 64];  // vT:  [d][s_local]
  __shared__ _Float16 sK[64 * 64];  // k'T: [m][s_local]
  const int bh = blockIdx.x;
  const int t = threadIdx.x;
  const int wave = t >> 5, lane = t & 31;
  const int r = lane & 15, hi = lane >> 4;
  const int wd = wave & 3;   // d tile (16 rows)
  const int wm2 = wave >> 2; // pair of m tiles (32 cols)
  const _Float16* kb = kprime + (size_t)bh * S_LEN * NF;
  const _Float16* vb = vmat + (size_t)bh * S_LEN * HD;

  v8f acc[2] = {};

  for (int s0 = 0; s0 < S_LEN; s0 += 64) {
#pragma unroll
    for (int p = 0; p < 2; ++p) {
      int i = (p * 256 + t) * 8;
      int sr = i >> 6, c = i & 63;
      v8h kv8 = *(const v8h*)&kb[(long)(s0 + sr) * NF + c];
      v8h vv8 = *(const v8h*)&vb[(long)(s0 + sr) * HD + c];
#pragma unroll
      for (int e = 0; e < 8; ++e) {
        sK[(c + e) * 64 + sr] = kv8[e];
        sV[(c + e) * 64 + sr] = vv8[e];
      }
    }
    __syncthreads();

#pragma unroll
    for (int ks = 0; ks < 2; ++ks) {
      const _Float16* pa = &sV[(wd * 16 + r) * 64 + ks * 32 + hi * 16];
      v16h a = combine16(*(const v8h*)pa, *(const v8h*)(pa + 8));
#pragma unroll
      for (int tn = 0; tn < 2; ++tn) {
        const _Float16* pb = &sK[((wm2 * 2 + tn) * 16 + r) * 64 + ks * 32 + hi * 16];
        v16h b = *(const v16h*)pb;
        acc[tn] = __builtin_amdgcn_wmma_f32_16x16x32_f16(
            false, a, false, b, (short)0, acc[tn], false, false);
      }
    }
    __syncthreads();
  }

#pragma unroll
  for (int tn = 0; tn < 2; ++tn) {
    int m = (wm2 * 2 + tn) * 16 + r;
#pragma unroll
    for (int v = 0; v < 8; ++v) {
      int d = wd * 16 + hi * 8 + v;
      kvT[(size_t)bh * HD * NF + (long)d * NF + m] = (_Float16)acc[tn][v];
    }
  }
}

__global__ void f32_to_f16(const float* __restrict__ in,
                           _Float16* __restrict__ out, long n) {
  long i = (long)blockIdx.x * blockDim.x + threadIdx.x;
  if (i < n) out[i] = (_Float16)in[i];
}

// ---------------------------------------------------------------------------

extern "C" void kernel_launch(void* const* d_in, const int* in_sizes, int n_in,
                              void* d_out, int out_size, void* d_ws,
                              size_t ws_size, hipStream_t stream) {
  (void)in_sizes; (void)n_in; (void)out_size; (void)ws_size;

  const float* x  = (const float*)d_in[0];
  const float* Wq = (const float*)d_in[1];
  const float* bq = (const float*)d_in[2];
  const float* Wk = (const float*)d_in[3];
  const float* bk = (const float*)d_in[4];
  const float* Wv = (const float*)d_in[5];
  const float* bv = (const float*)d_in[6];
  const float* Wo = (const float*)d_in[7];
  const float* bo = (const float*)d_in[8];
  const float* P  = (const float*)d_in[9];
  float* out = (float*)d_out;

  const long ROWS = 4L * S_LEN;          // 16384 (b,s) rows
  const long BHS  = 4L * NH * S_LEN;     // 262144 (b,h,s) rows
  const long NX   = ROWS * DIM;          // 16.7M elements

  // Workspace layout (bytes), f16 tensors. attn reuses the xb region
  // (x no longer needed after the QKV projections).
  char* ws = (char*)d_ws;
  const size_t MB = 1024 * 1024;
  _Float16* xb   = (_Float16*)(ws + 0);          // [16384,1024]      32MB
  _Float16* attn = (_Float16*)(ws + 0);          // [b,s,1024]        (reuse)
  _Float16* qb   = (_Float16*)(ws + 32 * MB);    // [b,h,s,64]        32MB
  _Float16* kb   = (_Float16*)(ws + 64 * MB);    // [b,h,s,64]        32MB
  _Float16* vbuf = (_Float16*)(ws + 96 * MB);    // [b,h,s,64]        32MB
  _Float16* qp   = (_Float16*)(ws + 128 * MB);   // [b,h,s,64] q'     32MB
  _Float16* kp   = (_Float16*)(ws + 160 * MB);   // [b,h,s,64] k'     32MB
  _Float16* wqb  = (_Float16*)(ws + 192 * MB);   // 2MB each
  _Float16* wkb  = (_Float16*)(ws + 194 * MB);
  _Float16* wvb  = (_Float16*)(ws + 196 * MB);
  _Float16* wob  = (_Float16*)(ws + 198 * MB);
  _Float16* pb   = (_Float16*)(ws + 200 * MB);   // [16,64,64]  128KB
  _Float16* kvT  = (_Float16*)(ws + 201 * MB);   // [64,64,64]  512KB

  // 1) Convert inputs to f16.
  f32_to_f16<<<(NX + 255) / 256, 256, 0, stream>>>(x, xb, NX);
  f32_to_f16<<<(DIM * DIM + 255) / 256, 256, 0, stream>>>(Wq, wqb, DIM * DIM);
  f32_to_f16<<<(DIM * DIM + 255) / 256, 256, 0, stream>>>(Wk, wkb, DIM * DIM);
  f32_to_f16<<<(DIM * DIM + 255) / 256, 256, 0, stream>>>(Wv, wvb, DIM * DIM);
  f32_to_f16<<<(DIM * DIM + 255) / 256, 256, 0, stream>>>(Wo, wob, DIM * DIM);
  f32_to_f16<<<(NH * NF * HD + 255) / 256, 256, 0, stream>>>(P, pb, NH * NF * HD);

  // 2) Q/K/V projections: [16384,1024] x W^T + b -> split-head f16.
  dim3 gBig(ROWS / 128, DIM / 64);
  gemm_wmma<1, false, true, true><<<gBig, 256, 0, stream>>>(xb, wqb, bq, qb, DIM, 0);
  gemm_wmma<1, false, true, true><<<gBig, 256, 0, stream>>>(xb, wkb, bk, kb, DIM, 0);
  gemm_wmma<1, false, true, true><<<gBig, 256, 0, stream>>>(xb, wvb, bv, vbuf, DIM, 0);

  // 3) Feature maps: relu(t @ P[h]^T) / sqrt(M), per-head B (mat_mod = H).
  dim3 gFeat(BHS / 128, NF / 64);
  gemm_wmma<0, true, false, true><<<gFeat, 256, 0, stream>>>(qb, pb, nullptr, qp, HD, NH);
  gemm_wmma<0, true, false, true><<<gFeat, 256, 0, stream>>>(kb, pb, nullptr, kp, HD, NH);

  // 4) kv^T[b,h][d][m] = sum_s v[s,d] k'[s,m].
  kv_wmma<<<4 * NH, 256, 0, stream>>>(kp, vbuf, kvT);

  // 5) attn = q' @ kv (B = kvT[n=d][k=m], per-(b,h) matrix: mat_mod = B*H),
  //    stored merged-head [b,s,1024].
  gemm_wmma<2, false, false, true><<<gFeat, 256, 0, stream>>>(qp, kvT, nullptr, attn, NF, 4 * NH);

  // 6) Output projection: attn @ Wo^T + bo -> f32 d_out.
  gemm_wmma<0, false, true, false><<<gBig, 256, 0, stream>>>(attn, wob, bo, out, DIM, 0);
}
